// GRUClassifier_36979668418855
// MI455X (gfx1250) — compile-verified, matrix operand order checked by reading, not run on hardware
//
#include <hip/hip_runtime.h>
#include <hip/hip_bf16.h>
#include <math.h>

// ---------------------------------------------------------------------------
// GRU classifier for MI455X (gfx1250, wave32).
//   Phase 1: f32 -> bf16 conversion of x, Wh, Wz.
//   Phase 2: xh = x@Wh^T, xz = x@Wz^T via v_wmma_f32_16x16x32_bf16
//            (each wave: 16x32 output tile, A-fragment reused across 2 WMMAs).
//   Phase 3: persistent-scan kernel, U-matrices staged into LDS (256KB/WGP)
//            with GLOBAL_LOAD_ASYNC_TO_LDS_B128; device-wide software barrier
//            per timestep.
//   Phase 4: logits + softmax (single block).
// Wr/Ur/br/xr are dead code in the reference and are skipped entirely.
// ---------------------------------------------------------------------------

#ifndef __has_builtin
#define __has_builtin(x) 0
#endif

#if __has_builtin(__builtin_amdgcn_global_load_async_to_lds_b128)
#define HAVE_ASYNC_LDS 1
#else
#define HAVE_ASYNC_LDS 0
#endif

typedef __attribute__((ext_vector_type(16))) __bf16 v16bf;
typedef __attribute__((ext_vector_type(8)))  __bf16 v8bf;
typedef __attribute__((ext_vector_type(8)))  float  v8f;
typedef __attribute__((ext_vector_type(4)))  float  v4f;
typedef __attribute__((ext_vector_type(4)))  int    v4i;

constexpr int kT    = 1024;
constexpr int kD    = 2048;
constexpr int kNH   = 2048;
constexpr int kCATS = 1000;

constexpr int NBLK = 128;          // persistent scan workgroups
constexpr int RPB  = kNH / NBLK;   // 16 rows per block

union ABfrag { v16bf v; v8bf h[2]; };

#if HAVE_ASYNC_LDS
// Builtin prototype (from the toolchain diagnostic):
//   void(v4i addrspace(1)*, v4i addrspace(3)*, imm int offset, imm int cpol)
typedef __attribute__((address_space(1))) v4i gv4i_t;
typedef __attribute__((address_space(3))) v4i sv4i_t;

__device__ __forceinline__ void async_copy_b128(const float* g, float* l) {
  __builtin_amdgcn_global_load_async_to_lds_b128(
      (gv4i_t*)const_cast<float*>(g), (sv4i_t*)l, 0, 0);
}
__device__ __forceinline__ void wait_async_zero() {
#if __has_builtin(__builtin_amdgcn_s_wait_asynccnt)
  __builtin_amdgcn_s_wait_asynccnt(0);
#else
  asm volatile("s_wait_asynccnt 0x0" ::: "memory");
#endif
}
#endif

// ---------------- Phase 1: f32 -> bf16 (round-to-nearest-even) -------------
__global__ void f32_to_bf16_kernel(const float* __restrict__ src,
                                   unsigned short* __restrict__ dst, int n) {
  int i = blockIdx.x * blockDim.x + threadIdx.x;
  if (i < n) {
    unsigned u = __float_as_uint(src[i]);
    unsigned r = (u + 0x7FFFu + ((u >> 16) & 1u)) >> 16;
    dst[i] = (unsigned short)r;
  }
}

// ---------------- Phase 2: WMMA projection GEMMs ---------------------------
// C[T,NH] = Xbf[T,D] * W^T, W stored [NH,D] row-major so B(k,n) = W[n,k]:
// each lane fetches 16B-contiguous K-groups from row n of W.
// Per-wave tile: 16(M) x 32(N); one A fragment feeds two WMMAs.
__global__ __launch_bounds__(256)
void proj_gemm_kernel(const unsigned short* __restrict__ Xbf,
                      const unsigned short* __restrict__ Whbf,
                      const unsigned short* __restrict__ Wzbf,
                      float* __restrict__ xh, float* __restrict__ xz) {
  const unsigned short* W = (blockIdx.z == 0) ? Whbf : Wzbf;
  float*                C = (blockIdx.z == 0) ? xh   : xz;

  const int lane = threadIdx.x & 31;
  const int wave = threadIdx.x >> 5;
  const int half = lane >> 4;       // K-half selector per ISA A/B layout
  const int sub  = lane & 15;       // row (A) / col (B) within 16x16 tile
  const int m0   = blockIdx.x * 32  + (wave >> 2) * 16;
  const int n0   = blockIdx.y * 128 + (wave & 3) * 32;

  const unsigned short* arow  = Xbf + (size_t)(m0 + sub) * kD;
  const unsigned short* brow0 = W   + (size_t)(n0 + sub) * kD;
  const unsigned short* brow1 = W   + (size_t)(n0 + 16 + sub) * kD;

  v8f acc0 = {}, acc1 = {};
  for (int k = 0; k < kD; k += 32) {
    const int kb = k + half * 8;
    ABfrag a, b0, b1;
    a.h[0]  = *reinterpret_cast<const v8bf*>(arow + kb);
    a.h[1]  = *reinterpret_cast<const v8bf*>(arow + kb + 16);
    b0.h[0] = *reinterpret_cast<const v8bf*>(brow0 + kb);
    b0.h[1] = *reinterpret_cast<const v8bf*>(brow0 + kb + 16);
    b1.h[0] = *reinterpret_cast<const v8bf*>(brow1 + kb);
    b1.h[1] = *reinterpret_cast<const v8bf*>(brow1 + kb + 16);
    __builtin_prefetch(arow + kb + 32, 0, 1);    // global_prefetch_b8
    __builtin_prefetch(brow0 + kb + 32, 0, 1);
    __builtin_prefetch(brow1 + kb + 32, 0, 1);
    // (neg_a, A, neg_b, B, c_mod, C, reuse_a, reuse_b)
    acc0 = __builtin_amdgcn_wmma_f32_16x16x32_bf16(
        false, a.v, false, b0.v, (short)0, acc0, false, false);
    acc1 = __builtin_amdgcn_wmma_f32_16x16x32_bf16(
        false, a.v, false, b1.v, (short)0, acc1, false, false);
  }
  // C/D layout: lanes 0-15 -> rows j, lanes 16-31 -> rows j+8 (VGPR j).
  float* crow0 = C + (size_t)(m0 + half * 8) * kNH + (n0 + sub);
  float* crow1 = crow0 + 16;
#pragma unroll
  for (int j = 0; j < 8; ++j) {
    crow0[(size_t)j * kNH] = acc0[j];
    crow1[(size_t)j * kNH] = acc1[j];
  }
}

// ---------------- Phase 3: persistent recurrent scan -----------------------
__global__ void init_state_kernel(const float* __restrict__ zt0,
                                  const float* __restrict__ ht0,
                                  const float* __restrict__ hp0,
                                  float* hp, float* zb, float* htb) {
  int i = blockIdx.x * blockDim.x + threadIdx.x;
  if (i < kNH) { hp[i] = hp0[i]; zb[i] = zt0[i]; htb[i] = ht0[i]; }
}

__global__ __launch_bounds__(256)
void gru_scan_kernel(const float* __restrict__ Uh, const float* __restrict__ Uz,
                     const float* __restrict__ bz,
                     const float* __restrict__ xh, const float* __restrict__ xz,
                     float* hp,            // ping-pong [2][kNH]
                     float* zbuf, float* htbuf,
                     unsigned* barrier_cnt) {
  extern __shared__ float lds[];
  float* sUz = lds;                          // RPB*kNH
  float* sUh = lds + (size_t)RPB * kNH;      // RPB*kNH
  float* sHp = lds + (size_t)2 * RPB * kNH;  // kNH

  const int tid  = threadIdx.x;
  const int row0 = blockIdx.x * RPB;

  // Stage this block's U slices into LDS once: 2 * 16 rows * 8KB = 256KB.
  {
    const float* gz = Uz + (size_t)row0 * kNH;
    const float* gh = Uh + (size_t)row0 * kNH;
    const int chunks = RPB * kNH / 4;  // 16B chunks per matrix
#if HAVE_ASYNC_LDS
    for (int i = tid; i < chunks; i += blockDim.x) {
      async_copy_b128(gz + (size_t)i * 4, sUz + (size_t)i * 4);
      async_copy_b128(gh + (size_t)i * 4, sUh + (size_t)i * 4);
    }
    wait_async_zero();
#else
    for (int i = tid; i < chunks; i += blockDim.x) {
      *reinterpret_cast<v4f*>(sUz + (size_t)i * 4) =
          *reinterpret_cast<const v4f*>(gz + (size_t)i * 4);
      *reinterpret_cast<v4f*>(sUh + (size_t)i * 4) =
          *reinterpret_cast<const v4f*>(gh + (size_t)i * 4);
    }
#endif
  }
  __syncthreads();

  const int wave = tid >> 5, lane = tid & 31;

  for (int t = 0; t < kT; ++t) {
    volatile const float* hpin = hp + (t & 1) * kNH;       // coherent read
    float* hpout = hp + ((t + 1) & 1) * kNH;

    // Broadcast h_prev into LDS (8KB).
    for (int i = tid; i < kNH; i += 256) sHp[i] = hpin[i];
    __syncthreads();

    // One wave per output row; lanes stride K; butterfly reduce (wave32).
    for (int rr = wave; rr < RPB; rr += 8) {
      const int gr = row0 + rr;
      const float* uz = sUz + (size_t)rr * kNH;
      const float* uh = sUh + (size_t)rr * kNH;
      float az = 0.f, ah = 0.f;
      for (int k = lane; k < kNH; k += 32) {
        float hv = sHp[k];
        az = fmaf(uz[k], hv, az);
        ah = fmaf(uh[k], hv, ah);
      }
      for (int off = 16; off > 0; off >>= 1) {
        az += __shfl_xor(az, off, 32);
        ah += __shfl_xor(ah, off, 32);
      }
      if (lane == 0) {
        // Stale-gate update order, faithful to the reference.
        float zo = zbuf[gr], hto = htbuf[gr], hpo = sHp[gr];
        hpout[gr] = (1.f - zo) * hpo + zo * hto;           // h_{t+1}
        float zarg = xz[(size_t)t * kNH + gr] + az + bz[gr];
        zbuf[gr]  = 1.f / (1.f + __expf(-zarg));           // z_{t+1}
        htbuf[gr] = tanhf(xh[(size_t)t * kNH + gr] + ah);  // htilde_{t+1}
      }
    }
    __syncthreads();

    // Device-wide barrier: monotonic arrival counter (max 1024*128 < 2^32).
    if (tid == 0) {
      __threadfence();
      atomicAdd(barrier_cnt, 1u);
      const unsigned target = (unsigned)(t + 1) * (unsigned)NBLK;
      while (__hip_atomic_load(barrier_cnt, __ATOMIC_RELAXED,
                               __HIP_MEMORY_SCOPE_AGENT) < target) {
        __builtin_amdgcn_s_sleep(2);
      }
    }
    __syncthreads();
  }
}

// ---------------- Phase 4: logits + softmax --------------------------------
__global__ __launch_bounds__(256)
void head_kernel(const float* __restrict__ h, const float* __restrict__ Wout,
                 float* __restrict__ out) {
  __shared__ float sl[kCATS];
  __shared__ float red[256];
  const int tid = threadIdx.x;

  for (int c = tid; c < kCATS; c += 256) {
    const float* w = Wout + (size_t)c * kNH;
    float acc = 0.f;
    for (int k = 0; k < kNH; ++k) acc = fmaf(w[k], h[k], acc);
    sl[c] = acc;
  }
  __syncthreads();

  float m = -3.4e38f;
  for (int c = tid; c < kCATS; c += 256) m = fmaxf(m, sl[c]);
  red[tid] = m; __syncthreads();
  for (int s = 128; s > 0; s >>= 1) {
    if (tid < s) red[tid] = fmaxf(red[tid], red[tid + s]);
    __syncthreads();
  }
  m = red[0]; __syncthreads();

  float ssum = 0.f;
  for (int c = tid; c < kCATS; c += 256) {
    float e = __expf(sl[c] - m);
    sl[c] = e; ssum += e;
  }
  red[tid] = ssum; __syncthreads();
  for (int s = 128; s > 0; s >>= 1) {
    if (tid < s) red[tid] += red[tid + s];
    __syncthreads();
  }
  const float inv = 1.f / red[0];
  for (int c = tid; c < kCATS; c += 256) out[c] = sl[c] * inv;
}

// ---------------------------------------------------------------------------
extern "C" void kernel_launch(void* const* d_in, const int* in_sizes, int n_in,
                              void* d_out, int out_size, void* d_ws, size_t ws_size,
                              hipStream_t stream) {
  (void)in_sizes; (void)n_in; (void)out_size; (void)ws_size;

  const float* x    = (const float*)d_in[0];
  const float* Wh   = (const float*)d_in[1];
  const float* Wz   = (const float*)d_in[2];
  // d_in[3] = Wr (dead)
  const float* Uh   = (const float*)d_in[4];
  const float* Uz   = (const float*)d_in[5];
  const float* bz   = (const float*)d_in[6];
  // d_in[7] = Ur, d_in[8] = br (dead)
  const float* Wout = (const float*)d_in[9];
  // d_in[10] = h0 (overwritten at step 0 before being observed; T>0)
  const float* zt0  = (const float*)d_in[11];
  const float* ht0  = (const float*)d_in[12];
  const float* hp0  = (const float*)d_in[13];

  // Workspace carve-up (~38 MB total).
  char* ws = (char*)d_ws;
  size_t off = 0;
  auto take = [&](size_t bytes) -> char* {
    char* p = ws + off;
    off = (off + bytes + 255) & ~(size_t)255;
    return p;
  };
  unsigned short* xbf  = (unsigned short*)take((size_t)kT  * kD * 2);
  unsigned short* whbf = (unsigned short*)take((size_t)kNH * kD * 2);
  unsigned short* wzbf = (unsigned short*)take((size_t)kNH * kD * 2);
  float* xh = (float*)take((size_t)kT * kNH * 4);
  float* xz = (float*)take((size_t)kT * kNH * 4);
  float* hp = (float*)take((size_t)2 * kNH * 4);
  float* zb = (float*)take((size_t)kNH * 4);
  float* htb = (float*)take((size_t)kNH * 4);
  unsigned* cnt = (unsigned*)take(256);

  (void)hipMemsetAsync(cnt, 0, sizeof(unsigned), stream);  // graph-capturable

  f32_to_bf16_kernel<<<(kT * kD + 255) / 256, 256, 0, stream>>>(x, xbf, kT * kD);
  f32_to_bf16_kernel<<<(kNH * kD + 255) / 256, 256, 0, stream>>>(Wh, whbf, kNH * kD);
  f32_to_bf16_kernel<<<(kNH * kD + 255) / 256, 256, 0, stream>>>(Wz, wzbf, kNH * kD);

  dim3 ggrid(kT / 32, kNH / 128, 2);  // 32x128 block tile, z selects Wh/Wz
  proj_gemm_kernel<<<ggrid, 256, 0, stream>>>(xbf, whbf, wzbf, xh, xz);

  init_state_kernel<<<(kNH + 255) / 256, 256, 0, stream>>>(zt0, ht0, hp0, hp, zb, htb);

  // 2*16 rows of U (f32) + h broadcast in LDS: 256KB + 8KB <= 320KB/WGP.
  size_t ldsBytes = ((size_t)2 * RPB * kNH + kNH) * sizeof(float);
  gru_scan_kernel<<<NBLK, 256, ldsBytes, stream>>>(Uh, Uz, bz, xh, xz, hp, zb, htb, cnt);

  // kT even -> final h lives in hp[0..kNH).
  head_kernel<<<1, 256, 0, stream>>>(hp, Wout, (float*)d_out);
}
